// CandidateVoteModule_V2_26688926777520
// MI455X (gfx1250) — matrix operand order repaired and checked by reading, not compiled, not run on hardware
//
#include <hip/hip_runtime.h>
#include <stdint.h>

typedef __attribute__((ext_vector_type(16))) _Float16     v16h;
typedef __attribute__((ext_vector_type(8)))  _Float16     v8h;
typedef __attribute__((ext_vector_type(8)))  float        v8f;
typedef __attribute__((ext_vector_type(4)))  unsigned int v4u;
typedef __attribute__((ext_vector_type(8)))  int          v8i;
typedef __attribute__((ext_vector_type(4)))  int          v4i;

#define THREADS 256

// ---------------------------------------------------------------- FPS
__global__ __launch_bounds__(THREADS)
void fps_kernel(const float* __restrict__ xyz, float* __restrict__ mind,
                int* __restrict__ inds, int N, int npoint) {
  __shared__ float s_val[THREADS];
  __shared__ int   s_idx[THREADS];
  __shared__ int   s_last;
  int tid = threadIdx.x;
  for (int i = tid; i < N; i += THREADS) mind[i] = 1e10f;
  if (tid == 0) s_last = 0;
  __syncthreads();
  for (int it = 0; it < npoint; ++it) {
    int last = s_last;
    if (tid == 0) inds[it] = last;
    float lx = xyz[last * 3 + 0], ly = xyz[last * 3 + 1], lz = xyz[last * 3 + 2];
    float bestv = -1.0f; int besti = 0;
    for (int i = tid; i < N; i += THREADS) {
      float dx = xyz[i * 3 + 0] - lx;
      float dy = xyz[i * 3 + 1] - ly;
      float dz = xyz[i * 3 + 2] - lz;
      float d = dx * dx + dy * dy + dz * dz;
      float m = fminf(mind[i], d);
      mind[i] = m;
      if (m > bestv) { bestv = m; besti = i; }
    }
    s_val[tid] = bestv; s_idx[tid] = besti;
    __syncthreads();
    for (int off = THREADS / 2; off > 0; off >>= 1) {
      if (tid < off) {
        float vo = s_val[tid + off]; int io = s_idx[tid + off];
        if (vo > s_val[tid] || (vo == s_val[tid] && io < s_idx[tid])) {
          s_val[tid] = vo; s_idx[tid] = io;
        }
      }
      __syncthreads();
    }
    if (tid == 0) s_last = s_idx[0];
    __syncthreads();
  }
}

// ---------------------------------------------------------------- gather xyz
__global__ __launch_bounds__(THREADS)
void gather_xyz_kernel(const float* __restrict__ xyz, const int* __restrict__ inds,
                       float* __restrict__ out, int S) {
  int s = blockIdx.x * THREADS + threadIdx.x;
  if (s >= S) return;
  int n = inds[s];
  out[s * 3 + 0] = xyz[n * 3 + 0];
  out[s * 3 + 1] = xyz[n * 3 + 1];
  out[s * 3 + 2] = xyz[n * 3 + 2];
}

// ---------------------------------------------------------------- ball query
__global__ __launch_bounds__(THREADS)
void ball_query_kernel(const float* __restrict__ xyz, const float* __restrict__ new_xyz,
                       int* __restrict__ idx, int N, int S, float r2, int ns) {
  int s = blockIdx.x * THREADS + threadIdx.x;
  if (s >= S) return;
  float qx = new_xyz[s * 3 + 0], qy = new_xyz[s * 3 + 1], qz = new_xyz[s * 3 + 2];
  int cnt = 0;
  for (int n = 0; n < N && cnt < ns; ++n) {
    float dx = xyz[n * 3 + 0] - qx;
    float dy = xyz[n * 3 + 1] - qy;
    float dz = xyz[n * 3 + 2] - qz;
    if (dx * dx + dy * dy + dz * dz < r2) { idx[s * ns + cnt] = n; ++cnt; }
  }
  int first = (cnt > 0) ? idx[s * ns] : 0;
  for (int j = cnt; j < ns; ++j) idx[s * ns + j] = first;
}

// ---------------------------------------------------------------- grouping -> f16 X matrix, M-major: X[m][k], row stride Kpad
__global__ __launch_bounds__(THREADS)
void group_kernel(const float* __restrict__ src_xyz, const float* __restrict__ new_xyz,
                  const int* __restrict__ idx, const float* __restrict__ feats,
                  _Float16* __restrict__ X, int feat_cols, int C, int Kpad,
                  int S, int ns, float inv_r) {
  int m = blockIdx.x * THREADS + threadIdx.x;
  int M = S * ns;
  if (m >= M) return;
  int s = m / ns;
  int n = idx[m];
  _Float16* row = X + (long)m * Kpad;
  float cx = new_xyz[s * 3 + 0], cy = new_xyz[s * 3 + 1], cz = new_xyz[s * 3 + 2];
  row[0] = (_Float16)((src_xyz[n * 3 + 0] - cx) * inv_r);
  row[1] = (_Float16)((src_xyz[n * 3 + 1] - cy) * inv_r);
  row[2] = (_Float16)((src_xyz[n * 3 + 2] - cz) * inv_r);
  for (int c = 3; c < C; ++c)
    row[c] = (_Float16)feats[(long)(c - 3) * feat_cols + n];
  for (int c = C; c < Kpad; ++c)
    row[c] = (_Float16)0.0f;
}

// ---------------------------------------------------------------- WMMA GEMM, TDM-staged LDS tiles
// Y(OxM) f32 = W(OxK) f32 * X(M x Kpad, M-major f16). Each wave: 16(O) x 32(M), 2 WMMAs sharing A.
__global__ __launch_bounds__(THREADS)
void wmma_gemm_kernel(const float* __restrict__ W, const _Float16* __restrict__ X,
                      float* __restrict__ Y, int O, int K, int Kpad, int M) {
  __shared__ _Float16 Xlds[128 * 32];   // 128 m-rows x 32 k  (8 KB)
  __shared__ _Float16 Wlds[32 * 32];    // 32 o-rows  x 32 k  (2 KB)
  int tid = threadIdx.x;
  int lane = tid & 31, wave = tid >> 5;
  int wave_o = wave >> 2;               // 0..1 -> 16-row slabs of O tile
  int wave_m = wave & 3;                // 0..3 -> 32-col slabs of M tile
  int m0 = blockIdx.x * 128;
  int o0 = blockIdx.y * 32;
  int l15 = lane & 15;
  int hi  = lane >> 4;
  unsigned lds_base = (unsigned)(unsigned long long)(void*)&Xlds[0];
  unsigned long long gbase = (unsigned long long)(const void*)X;
  v8f acc0 = {}, acc1 = {};
  union F16x16 { v16h v; v8h h[2]; };
  for (int k0 = 0; k0 < Kpad; k0 += 32) {
    if (tid < 32) {
      // TDM: 2D tile, elem=2B, tile 32(k,x) x 128(m,y rows), row stride = Kpad
      unsigned long long ga = gbase + ((unsigned long long)m0 * (unsigned long long)Kpad +
                                       (unsigned long long)k0) * 2ull;
      unsigned td0 = (unsigned)Kpad;         // tensor width (elements)
      unsigned td1 = (unsigned)(M - m0);     // remaining rows
      v4u g0; v8i g1; v4i g2 = {0, 0, 0, 0}; v4i g3 = {0, 0, 0, 0};
      v8i g4 = {0, 0, 0, 0, 0, 0, 0, 0};
      g0[0] = 1u;                                      // count=1 (valid), user mode
      g0[1] = lds_base;                                // lds_addr
      g0[2] = (unsigned)(ga & 0xffffffffu);            // global_addr[31:0]
      g0[3] = (unsigned)((ga >> 32) & 0x01ffffffu) | (2u << 30);  // addr[56:32] | type=2
      g1[0] = (int)(1u << 16);                         // data_size=1 -> 2 bytes
      g1[1] = (int)((td0 & 0xffffu) << 16);            // tensor_dim0[15:0]
      g1[2] = (int)((td0 >> 16) | ((td1 & 0xffffu) << 16));  // td0[31:16] | td1[15:0]
      g1[3] = (int)((td1 >> 16) | (32u << 16));        // td1[31:16] | tile_dim0=32
      g1[4] = (int)128u;                               // tile_dim1=128, tile_dim2=0
      g1[5] = (int)td0;                                // tensor_dim0_stride = Kpad
      g1[6] = 0;
      g1[7] = 0;
      __builtin_amdgcn_tensor_load_to_lds(g0, g1, g2, g3, g4, 0);
      __builtin_amdgcn_s_wait_tensorcnt(0);
    }
    // stage W chunk (f32 -> f16), zero-padded outside [O,K)
    for (int e = tid; e < 32 * 32; e += THREADS) {
      int oo = e >> 5, kk = e & 31;
      float w = 0.0f;
      if ((o0 + oo) < O && (k0 + kk) < K) w = W[(long)(o0 + oo) * K + (k0 + kk)];
      Wlds[e] = (_Float16)w;
    }
    __syncthreads();
    // A fragment (ISA 7.12.2): lane row o, k runs {base..base+7, base+16..base+23}, base = hi?8:0
    F16x16 ua;
    {
      const _Float16* wrow = &Wlds[(wave_o * 16 + l15) * 32 + (hi ? 8 : 0)];
      ua.h[0] = *(const v8h*)(wrow);
      ua.h[1] = *(const v8h*)(wrow + 16);
    }
    // B fragments: lane col m, k runs {base..base+7, base+16..base+23}, base = hi?8:0
    F16x16 ub0, ub1;
    {
      const _Float16* x0 = &Xlds[(wave_m * 32 + l15) * 32 + (hi ? 8 : 0)];
      const _Float16* x1 = x0 + 16 * 32;
      ub0.h[0] = *(const v8h*)(x0);
      ub0.h[1] = *(const v8h*)(x0 + 16);
      ub1.h[0] = *(const v8h*)(x1);
      ub1.h[1] = *(const v8h*)(x1 + 16);
    }
    acc0 = __builtin_amdgcn_wmma_f32_16x16x32_f16(false, ua.v, false, ub0.v, (short)0,
                                                  acc0, false, false);
    acc1 = __builtin_amdgcn_wmma_f32_16x16x32_f16(false, ua.v, false, ub1.v, (short)0,
                                                  acc1, false, false);
    __syncthreads();
  }
  int ncol0 = m0 + wave_m * 32 + l15;
  int ncol1 = ncol0 + 16;
  int orow0 = o0 + wave_o * 16 + (hi ? 8 : 0);
#pragma unroll
  for (int r = 0; r < 8; ++r) {
    int o = orow0 + r;
    if (o < O) {
      Y[(long)o * M + ncol0] = acc0[r];
      Y[(long)o * M + ncol1] = acc1[r];
    }
  }
}

// ---------------------------------------------------------------- per-channel BN stats
__global__ __launch_bounds__(THREADS)
void colstats_kernel(const float* __restrict__ Y, float* __restrict__ stats, int M) {
  __shared__ float ss[THREADS], sq[THREADS];
  int o = blockIdx.x;
  const float* row = Y + (long)o * M;
  float a = 0.0f, b = 0.0f;
  for (int m = threadIdx.x; m < M; m += THREADS) { float v = row[m]; a += v; b += v * v; }
  ss[threadIdx.x] = a; sq[threadIdx.x] = b;
  __syncthreads();
  for (int off = THREADS / 2; off > 0; off >>= 1) {
    if (threadIdx.x < off) { ss[threadIdx.x] += ss[threadIdx.x + off];
                             sq[threadIdx.x] += sq[threadIdx.x + off]; }
    __syncthreads();
  }
  if (threadIdx.x == 0) {
    float mean = ss[0] / (float)M;
    float var = sq[0] / (float)M - mean * mean;
    stats[o * 2 + 0] = mean;
    stats[o * 2 + 1] = rsqrtf(fmaxf(var, 0.0f) + 1e-5f);
  }
}

// ---------------------------------------------------------------- BN + ReLU; f16 out is M-major (row stride O)
__global__ __launch_bounds__(THREADS)
void bn_relu_kernel(const float* __restrict__ Y, const float* __restrict__ stats,
                    const float* __restrict__ g, const float* __restrict__ be,
                    _Float16* __restrict__ Xout, float* __restrict__ f32out,
                    int O, int M) {
  long i = (long)blockIdx.x * THREADS + threadIdx.x;
  long tot = (long)O * M;
  if (i >= tot) return;
  int o = (int)(i % O);       // Xout write is contiguous: i = m*O + o
  long m = i / O;
  float v = (Y[(long)o * M + m] - stats[o * 2]) * stats[o * 2 + 1] * g[o] + be[o];
  v = fmaxf(v, 0.0f);
  Xout[i] = (_Float16)v;
  if (f32out) f32out[(long)o * M + m] = v;
}

// ---------------------------------------------------------------- max over nsample (X is M-major, row stride O)
__global__ __launch_bounds__(THREADS)
void maxpool_kernel(const _Float16* __restrict__ Xin, float* __restrict__ f,
                    int O, int S, int ns) {
  int i = blockIdx.x * THREADS + threadIdx.x;
  if (i >= O * S) return;
  int o = i % O, s = i / O;
  const _Float16* p = Xin + (long)s * ns * O + o;
  float m = -1e30f;
  for (int j = 0; j < ns; ++j) m = fmaxf(m, (float)p[(long)j * O]);
  f[(long)o * S + s] = m;
}

// ---------------------------------------------------------------- 3-NN search + weights
__global__ __launch_bounds__(THREADS)
void three_nn_kernel(const float* __restrict__ unknown, const float* __restrict__ known,
                     int Nu, int Nk, int* __restrict__ nidx, float* __restrict__ nw) {
  int i = blockIdx.x * THREADS + threadIdx.x;
  if (i >= Nu) return;
  float ux = unknown[i * 3 + 0], uy = unknown[i * 3 + 1], uz = unknown[i * 3 + 2];
  float d0 = 1e30f, d1 = 1e30f, d2 = 1e30f;
  int i0 = 0, i1 = 0, i2 = 0;
  for (int k = 0; k < Nk; ++k) {
    float dx = known[k * 3 + 0] - ux;
    float dy = known[k * 3 + 1] - uy;
    float dz = known[k * 3 + 2] - uz;
    float d = dx * dx + dy * dy + dz * dz;
    if (d < d0)      { d2 = d1; i2 = i1; d1 = d0; i1 = i0; d0 = d; i0 = k; }
    else if (d < d1) { d2 = d1; i2 = i1; d1 = d;  i1 = k; }
    else if (d < d2) { d2 = d;  i2 = k; }
  }
  float w0 = 1.0f / (fmaxf(d0, 0.0f) + 1e-8f);
  float w1 = 1.0f / (fmaxf(d1, 0.0f) + 1e-8f);
  float w2 = 1.0f / (fmaxf(d2, 0.0f) + 1e-8f);
  float wsum = w0 + w1 + w2;
  nidx[i * 3 + 0] = i0; nidx[i * 3 + 1] = i1; nidx[i * 3 + 2] = i2;
  nw[i * 3 + 0] = w0 / wsum; nw[i * 3 + 1] = w1 / wsum; nw[i * 3 + 2] = w2 / wsum;
}

// ---------------------------------------------------------------- FP: interpolate + concat -> f16 X (M-major, row stride Ck+Cu)
__global__ __launch_bounds__(THREADS)
void fp_concat_kernel(const float* __restrict__ known_f, const float* __restrict__ unk_f,
                      const int* __restrict__ nidx, const float* __restrict__ nw,
                      _Float16* __restrict__ X, int Ck, int Cu, int Nu, int Nk) {
  long i = (long)blockIdx.x * THREADS + threadIdx.x;
  int Kp = Ck + Cu;
  long total = (long)Kp * Nu;
  if (i >= total) return;
  int c = (int)(i % Kp);
  int n = (int)(i / Kp);
  float v;
  if (c < Ck) {
    const float* row = known_f + (long)c * Nk;
    v = nw[n * 3 + 0] * row[nidx[n * 3 + 0]] +
        nw[n * 3 + 1] * row[nidx[n * 3 + 1]] +
        nw[n * 3 + 2] * row[nidx[n * 3 + 2]];
  } else {
    v = unk_f[(long)(c - Ck) * Nu + n];
  }
  X[i] = (_Float16)v;   // i == n*Kp + c : contiguous
}

// ---------------------------------------------------------------- vote head epilogue
__global__ __launch_bounds__(THREADS)
void vote_head_kernel(const float* __restrict__ Y3, const float* __restrict__ b3,
                      const float* __restrict__ xyz, const float* __restrict__ seed,
                      float* __restrict__ out_vxyz, float* __restrict__ out_off,
                      float* __restrict__ out_score, float* __restrict__ out_vf,
                      int N, int Cf) {
  int n = blockIdx.x * THREADS + threadIdx.x;
  if (n >= N) return;
  float ox = Y3[0 * (long)N + n] + b3[0];
  float oy = Y3[1 * (long)N + n] + b3[1];
  float oz = Y3[2 * (long)N + n] + b3[2];
  out_off[n * 3 + 0] = ox; out_off[n * 3 + 1] = oy; out_off[n * 3 + 2] = oz;
  out_vxyz[n * 3 + 0] = xyz[n * 3 + 0] + ox;
  out_vxyz[n * 3 + 1] = xyz[n * 3 + 1] + oy;
  out_vxyz[n * 3 + 2] = xyz[n * 3 + 2] + oz;
  float sc = Y3[3 * (long)N + n] + b3[3];
  out_score[n] = 1.0f / (1.0f + __expf(-sc));
  float nrm = 0.0f;
  for (int c = 0; c < Cf; ++c) {
    float v = seed[(long)c * N + n] + Y3[(long)(4 + c) * N + n] + b3[4 + c];
    nrm += v * v;
  }
  nrm = fmaxf(sqrtf(nrm), 1e-12f);
  for (int c = 0; c < Cf; ++c) {
    float v = seed[(long)c * N + n] + Y3[(long)(4 + c) * N + n] + b3[4 + c];
    out_vf[(long)n * Cf + c] = v / nrm;
  }
}

// ================================================================ host
extern "C" void kernel_launch(void* const* d_in, const int* in_sizes, int n_in,
                              void* d_out, int out_size, void* d_ws, size_t ws_size,
                              hipStream_t stream) {
  (void)in_sizes; (void)n_in; (void)out_size; (void)ws_size;
  const int B = 2, N = 16384, S1 = 2048, S2 = 1024, NS = 32, CF = 128;
  const int M1 = S1 * NS, M2 = S2 * NS;

  const float* xyz_all  = (const float*)d_in[0];
  const float* feat_all = (const float*)d_in[1];
  const float *saW[2][3], *saG[2][3], *saB[2][3];
  int pi = 2;
  for (int s = 0; s < 2; ++s)
    for (int l = 0; l < 3; ++l) {
      saW[s][l] = (const float*)d_in[pi + 0];
      saG[s][l] = (const float*)d_in[pi + 1];
      saB[s][l] = (const float*)d_in[pi + 2];
      pi += 3;
    }
  const float *fpW[2][2], *fpG[2][2], *fpB[2][2];
  for (int s = 0; s < 2; ++s)
    for (int l = 0; l < 2; ++l) {
      fpW[s][l] = (const float*)d_in[pi + 0];
      fpG[s][l] = (const float*)d_in[pi + 1];
      fpB[s][l] = (const float*)d_in[pi + 2];
      pi += 3;
    }
  const float* vW1 = (const float*)d_in[32];
  const float* vG1 = (const float*)d_in[34];
  const float* vE1 = (const float*)d_in[35];
  const float* vW2 = (const float*)d_in[36];
  const float* vG2 = (const float*)d_in[38];
  const float* vE2 = (const float*)d_in[39];
  const float* vW3 = (const float*)d_in[40];
  const float* vB3 = (const float*)d_in[41];

  // workspace carve-up (256B aligned)
  char* base = (char*)d_ws;
  size_t off = 0;
  auto carve = [&](size_t bytes) -> char* {
    char* p = base + off;
    off = (off + bytes + 255) & ~(size_t)255;
    return p;
  };
  _Float16* Xbuf  = (_Float16*)carve((size_t)256 * M1 * 2);   // 32 MB (max M*Kpad)
  float*    Ybuf  = (float*)   carve((size_t)256 * M1 * 4);   // 64 MB
  float*    mind  = (float*)   carve((size_t)N * 4);
  int*      inds1 = (int*)     carve((size_t)S1 * 4);
  int*      inds2 = (int*)     carve((size_t)S2 * 4);
  float*    xyz1  = (float*)   carve((size_t)S1 * 3 * 4);
  float*    xyz2  = (float*)   carve((size_t)S2 * 3 * 4);
  int*      idx1  = (int*)     carve((size_t)S1 * NS * 4);
  int*      idx2  = (int*)     carve((size_t)S2 * NS * 4);
  float*    f1    = (float*)   carve((size_t)256 * S1 * 4);
  float*    f2    = (float*)   carve((size_t)256 * S2 * 4);
  float*    fpo1  = (float*)   carve((size_t)256 * S1 * 4);
  float*    seed  = (float*)   carve((size_t)CF * N * 4);
  int*      nidx  = (int*)     carve((size_t)N * 3 * 4);
  float*    nw    = (float*)   carve((size_t)N * 3 * 4);
  float*    stats = (float*)   carve((size_t)512 * 4);

  float* out = (float*)d_out;
  float* out_vxyz  = out;
  float* out_off   = out + (size_t)B * N * 3;
  float* out_score = out + (size_t)2 * B * N * 3;
  float* out_vf    = out + (size_t)2 * B * N * 3 + (size_t)B * N;

  auto run_layer = [&](const float* W, const float* g, const float* be,
                       int O, int K, int Kpad, int M, float* f32out) {
    int Opad = (O + 31) & ~31;
    wmma_gemm_kernel<<<dim3(M / 128, Opad / 32), THREADS, 0, stream>>>(
        W, Xbuf, Ybuf, O, K, Kpad, M);
    colstats_kernel<<<O, THREADS, 0, stream>>>(Ybuf, stats, M);
    long tot = (long)O * M;
    bn_relu_kernel<<<(unsigned)((tot + THREADS - 1) / THREADS), THREADS, 0, stream>>>(
        Ybuf, stats, g, be, Xbuf, f32out, O, M);
  };

  for (int b = 0; b < B; ++b) {
    const float* xyz_b  = xyz_all + (size_t)b * N * 3;
    const float* feat_b = feat_all + (size_t)b * CF * N;

    // ---- SA1 ----
    fps_kernel<<<1, THREADS, 0, stream>>>(xyz_b, mind, inds1, N, S1);
    gather_xyz_kernel<<<(S1 + THREADS - 1) / THREADS, THREADS, 0, stream>>>(xyz_b, inds1, xyz1, S1);
    ball_query_kernel<<<(S1 + THREADS - 1) / THREADS, THREADS, 0, stream>>>(
        xyz_b, xyz1, idx1, N, S1, 0.2f * 0.2f, NS);
    group_kernel<<<M1 / THREADS, THREADS, 0, stream>>>(
        xyz_b, xyz1, idx1, feat_b, Xbuf, N, 3 + CF, 160, S1, NS, 1.0f / 0.2f);
    run_layer(saW[0][0], saG[0][0], saB[0][0], 128, 131, 160, M1, nullptr);
    run_layer(saW[0][1], saG[0][1], saB[0][1], 128, 128, 128, M1, nullptr);
    run_layer(saW[0][2], saG[0][2], saB[0][2], 256, 128, 128, M1, nullptr);
    maxpool_kernel<<<(256 * S1) / THREADS, THREADS, 0, stream>>>(Xbuf, f1, 256, S1, NS);

    // ---- SA2 ----
    fps_kernel<<<1, THREADS, 0, stream>>>(xyz1, mind, inds2, S1, S2);
    gather_xyz_kernel<<<(S2 + THREADS - 1) / THREADS, THREADS, 0, stream>>>(xyz1, inds2, xyz2, S2);
    ball_query_kernel<<<(S2 + THREADS - 1) / THREADS, THREADS, 0, stream>>>(
        xyz1, xyz2, idx2, S1, S2, 0.4f * 0.4f, NS);
    group_kernel<<<M2 / THREADS, THREADS, 0, stream>>>(
        xyz1, xyz2, idx2, f1, Xbuf, S1, 3 + 256, 288, S2, NS, 1.0f / 0.4f);
    run_layer(saW[1][0], saG[1][0], saB[1][0], 128, 259, 288, M2, nullptr);
    run_layer(saW[1][1], saG[1][1], saB[1][1], 128, 128, 128, M2, nullptr);
    run_layer(saW[1][2], saG[1][2], saB[1][2], 256, 128, 128, M2, nullptr);
    maxpool_kernel<<<(256 * S2) / THREADS, THREADS, 0, stream>>>(Xbuf, f2, 256, S2, NS);

    // ---- FP1: xyz1 <- xyz2 ----
    three_nn_kernel<<<(S1 + THREADS - 1) / THREADS, THREADS, 0, stream>>>(xyz1, xyz2, S1, S2, nidx, nw);
    {
      long tot = (long)512 * S1;
      fp_concat_kernel<<<(unsigned)((tot + THREADS - 1) / THREADS), THREADS, 0, stream>>>(
          f2, f1, nidx, nw, Xbuf, 256, 256, S1, S2);
    }
    run_layer(fpW[0][0], fpG[0][0], fpB[0][0], 256, 512, 512, S1, nullptr);
    run_layer(fpW[0][1], fpG[0][1], fpB[0][1], 256, 256, 256, S1, fpo1);

    // ---- FP2: xyz <- xyz1 ----
    three_nn_kernel<<<(N + THREADS - 1) / THREADS, THREADS, 0, stream>>>(xyz_b, xyz1, N, S1, nidx, nw);
    {
      long tot = (long)384 * N;
      fp_concat_kernel<<<(unsigned)((tot + THREADS - 1) / THREADS), THREADS, 0, stream>>>(
          fpo1, feat_b, nidx, nw, Xbuf, 256, CF, N, S1);
    }
    run_layer(fpW[1][0], fpG[1][0], fpB[1][0], 256, 384, 384, N, nullptr);
    run_layer(fpW[1][1], fpG[1][1], fpB[1][1], 128, 256, 256, N, seed);  // seed_features

    // ---- Vote head (b1/b2 cancel inside BN; only b3 applied) ----
    run_layer(vW1, vG1, vE1, 256, 128, 128, N, nullptr);
    run_layer(vW2, vG2, vE2, 256, 256, 256, N, nullptr);
    {
      int O = 132, Opad = 160;
      wmma_gemm_kernel<<<dim3(N / 128, Opad / 32), THREADS, 0, stream>>>(
          vW3, Xbuf, Ybuf, O, 256, 256, N);
    }
    vote_head_kernel<<<(N + THREADS - 1) / THREADS, THREADS, 0, stream>>>(
        Ybuf, vB3, xyz_b, seed,
        out_vxyz + (size_t)b * N * 3, out_off + (size_t)b * N * 3,
        out_score + (size_t)b * N, out_vf + (size_t)b * N * CF, N, CF);
  }
}
